// SelfNonParametric_70531952935530
// MI455X (gfx1250) — matrix operand order repaired and checked by reading, not compiled, not run on hardware
//
#include <hip/hip_runtime.h>

typedef __attribute__((ext_vector_type(16))) _Float16 v16h;
typedef __attribute__((ext_vector_type(8)))  _Float16 v8h;
typedef __attribute__((ext_vector_type(8)))  float    v8f;
typedef __attribute__((ext_vector_type(4)))  unsigned v4u;

#define N_TOTAL 16384
#define NB_LB   4096
#define NB_ULB  6144
#define DIM     128
#define NCLS    10
#define CHUNK   64
#define NCHUNKS (N_TOTAL / CHUNK)
#define TWO_LOG2E 2.8853900817779268f   // 2 / ln(2): exp(2x) == exp2(x * TWO_LOG2E)

// workspace layout (bytes)
#define OFF_FEATS 0                                   // 16384*128 f16 = 4 MB
#define OFF_VT    (N_TOTAL * DIM * 2)                 // 16*16384 f16 = 512 KB
#define OFF_CNT   (OFF_VT + 16 * N_TOTAL * 2)         // 10 f32 (padded)
#define OFF_DEN   (OFF_CNT + 64)                      // 10 f32

// ---------------------------------------------------------------- init
__global__ void init_kernel(float* __restrict__ cnt) {
    if (threadIdx.x < NCLS) cnt[threadIdx.x] = 0.0f;
}

// ---------------------------------------------------------------- labeled prep
__global__ void prep_lb_kernel(const float* __restrict__ oh,
                               _Float16* __restrict__ VT,
                               float* __restrict__ cnt) {
    int i = blockIdx.x * blockDim.x + threadIdx.x;
    if (i >= NB_LB) return;
#pragma unroll
    for (int c = 0; c < NCLS; ++c) {
        float v = oh[i * NCLS + c];
        VT[c * N_TOTAL + i] = (_Float16)v;
        if (v != 0.0f) atomicAdd(&cnt[c], v);
    }
    VT[10 * N_TOTAL + i] = (_Float16)1.0f;    // ones column -> row-sum
#pragma unroll
    for (int c = 11; c < 16; ++c) VT[c * N_TOTAL + i] = (_Float16)0.0f;
}

// ---------------------------------------------------------------- unlabeled prep
__global__ void prep_ulb_kernel(const float* __restrict__ lg1,
                                const float* __restrict__ lg2,
                                _Float16* __restrict__ VT,
                                float* __restrict__ cnt) {
    int i = blockIdx.x * blockDim.x + threadIdx.x;
    if (i >= NB_ULB) return;
    float a[NCLS], b[NCLS], g[NCLS];
    float m1 = -1e30f, m2 = -1e30f;
#pragma unroll
    for (int c = 0; c < NCLS; ++c) {
        a[c] = lg1[i * NCLS + c]; m1 = fmaxf(m1, a[c]);
        b[c] = lg2[i * NCLS + c]; m2 = fmaxf(m2, b[c]);
    }
    float s1 = 0.f, s2 = 0.f;
#pragma unroll
    for (int c = 0; c < NCLS; ++c) {
        s1 += __expf(2.0f * (a[c] - m1));
        s2 += __expf(2.0f * (b[c] - m2));
    }
    bool take1 = (s1 <= s2);                 // max(softmax1) >= max(softmax2)
    float mg = -1e30f;
#pragma unroll
    for (int c = 0; c < NCLS; ++c) { g[c] = take1 ? a[c] : b[c]; mg = fmaxf(mg, g[c]); }
    float se = 0.f;
#pragma unroll
    for (int c = 0; c < NCLS; ++c) se += __expf(g[c] - mg);
#pragma unroll
    for (int c = 0; c < NCLS; ++c) {
        float p = __expf(g[c] - mg) / se;    // temp-1 softmax for the cutoff mask
        if (p < 0.95f) g[c] = 0.0f;
    }
    float mv = g[0]; int mi = 0;
#pragma unroll
    for (int c = 1; c < NCLS; ++c) if (g[c] > mv) { mv = g[c]; mi = c; }
    if (mv != 0.0f) atomicAdd(&cnt[mi], 2.0f);   // counted twice (concat dup)
    float mm = -1e30f;
#pragma unroll
    for (int c = 0; c < NCLS; ++c) mm = fmaxf(mm, g[c]);
    float e[NCLS]; float ss = 0.f;
#pragma unroll
    for (int c = 0; c < NCLS; ++c) { e[c] = __expf(2.0f * (g[c] - mm)); ss += e[c]; }
    int r1 = NB_LB + i, r2 = NB_LB + NB_ULB + i;
#pragma unroll
    for (int c = 0; c < 16; ++c) {
        _Float16 h;
        if (c < NCLS)      h = (_Float16)(e[c] / ss);
        else if (c == 10)  h = (_Float16)1.0f;
        else               h = (_Float16)0.0f;
        VT[c * N_TOTAL + r1] = h;
        VT[c * N_TOTAL + r2] = h;
    }
}

// ---------------------------------------------------------------- L2 normalize -> f16
__global__ void norm_kernel(const float* __restrict__ lb,
                            const float* __restrict__ anc,
                            const float* __restrict__ pos,
                            _Float16* __restrict__ feats) {
    const int lane = threadIdx.x & 31;
    const int row  = blockIdx.x * 8 + (threadIdx.x >> 5);
    const float* src;
    if (row < NB_LB)                src = lb  + (size_t)row * DIM;
    else if (row < NB_LB + NB_ULB)  src = anc + (size_t)(row - NB_LB) * DIM;
    else                            src = pos + (size_t)(row - NB_LB - NB_ULB) * DIM;
    float4 v = reinterpret_cast<const float4*>(src)[lane];
    float s = v.x * v.x + v.y * v.y + v.z * v.z + v.w * v.w;
#pragma unroll
    for (int off = 16; off >= 1; off >>= 1) s += __shfl_xor(s, off);
    float r = 1.0f / fmaxf(sqrtf(s), 1e-12f);
    _Float16* dst = feats + (size_t)row * DIM + lane * 4;
    dst[0] = (_Float16)(v.x * r);
    dst[1] = (_Float16)(v.y * r);
    dst[2] = (_Float16)(v.z * r);
    dst[3] = (_Float16)(v.w * r);
}

// ---------------------------------------------------------------- class denominators
__global__ void finalize_kernel(const float* __restrict__ cnt, float* __restrict__ den) {
    if (threadIdx.x == 0) {
        float tot = 0.f;
        for (int c = 0; c < NCLS; ++c) tot += cnt[c];
        for (int c = 0; c < NCLS; ++c) {
            float d = (cnt[c] == 0.0f) ? tot : cnt[c];
            den[c] = 1.0f / d;
        }
    }
}

// ---------------------------------------------------------------- async helpers
// One 16-byte DMA: global (saddr base + vgpr byte offset) -> LDS, no VGPR staging.
__device__ __forceinline__ void async_ld16(unsigned ldsAddr, const _Float16* gbase,
                                           unsigned gByteOff) {
    asm volatile("global_load_async_to_lds_b128 %0, %1, %2"
                 :: "v"(ldsAddr), "v"(gByteOff),
                    "s"((unsigned long long)(uintptr_t)gbase)
                 : "memory");
}

// Stage one 64-column chunk (64x128 f16 feats + 16x64 f16 V^T) = 9 async b128/thread
__device__ __forceinline__ void stage_chunk(int chunk, int tid,
                                            const _Float16* feats, const _Float16* VT,
                                            unsigned ldsFbase, unsigned ldsVbase) {
#pragma unroll
    for (int it = 0; it < 8; ++it) {
        unsigned i4 = (unsigned)(tid + it * 128);
        async_ld16(ldsFbase + i4 * 16u, feats, (unsigned)chunk * (DIM * 2u) + i4 * 16u);
    }
    unsigned c = (unsigned)tid >> 3, j = (unsigned)tid & 7;
    async_ld16(ldsVbase + c * (CHUNK * 2u) + j * 16u, VT,
               (c * (unsigned)N_TOTAL + (unsigned)chunk) * 2u + j * 16u);
}

// ---------------------------------------------------------------- fused attention
// Per wave: one 16-row tile.  S^T = K_cols * Q^T via wmma (A = col feats,
// B = Q^T = contiguous row loads).  P^T assembled in registers (half-lane swap)
// and fed as B into O^T = V^T * P^T, accumulating numerator + row-sum together.
// Column chunks are double-buffered in LDS via global_load_async_to_lds_b128.
__global__ __launch_bounds__(128)
void attn_kernel(const _Float16* __restrict__ feats,
                 const _Float16* __restrict__ VT,
                 const float* __restrict__ den,
                 float* __restrict__ out) {
    __shared__ _Float16 ldsF[2][CHUNK * DIM];   // 2 x 16 KB
    __shared__ _Float16 ldsV[2][16 * CHUNK];    // 2 x  2 KB

    const int  tid  = threadIdx.x;
    const int  lane = tid & 31;
    const int  wave = tid >> 5;
    const int  l15  = lane & 15;
    const bool lo16 = (lane < 16);
    const int  koff = lo16 ? 0 : 8;
    const int  mRow = blockIdx.x * 64 + wave * 16 + l15;

    const unsigned ldsF0 = (unsigned)(uintptr_t)&ldsF[0][0];
    const unsigned ldsF1 = (unsigned)(uintptr_t)&ldsF[1][0];
    const unsigned ldsV0 = (unsigned)(uintptr_t)&ldsV[0][0];
    const unsigned ldsV1 = (unsigned)(uintptr_t)&ldsV[1][0];

    // Q^T as WMMA B-operand: lane m<16 holds K=0..7,16..23, m+16 holds 8..15,24..31
    v16h q[4];
#pragma unroll
    for (int kb = 0; kb < 4; ++kb) {
        const _Float16* p = feats + (size_t)mRow * DIM + kb * 32 + koff;
        v8h lo = *reinterpret_cast<const v8h*>(p);
        v8h hi = *reinterpret_cast<const v8h*>(p + 16);
        q[kb] = __builtin_shufflevector(lo, hi, 0,1,2,3,4,5,6,7,8,9,10,11,12,13,14,15);
    }

    v8f oacc = {0.f,0.f,0.f,0.f,0.f,0.f,0.f,0.f};

    // prologue: start DMA of chunk 0 into buffer 0
    stage_chunk(0, tid, feats, VT, ldsF0, ldsV0);

    for (int i = 0; i < NCHUNKS; ++i) {
        const int cur = i & 1;
        if (i + 1 < NCHUNKS) {
            // start DMA of next chunk into the other buffer (freed by the
            // trailing barrier of iteration i-1), then wait for chunk i only
            stage_chunk(i + 1, tid, feats, VT,
                        cur ? ldsF0 : ldsF1, cur ? ldsV0 : ldsV1);
            asm volatile("s_wait_asynccnt 0x9" ::: "memory");   // 9 newest may remain
        } else {
            asm volatile("s_wait_asynccnt 0x0" ::: "memory");
        }
        __syncthreads();                                        // all waves' slices landed

        const _Float16* F  = ldsF[cur];
        const _Float16* Vt = ldsV[cur];

#pragma unroll
        for (int g = 0; g < 2; ++g) {            // two 32-column groups
            // load BOTH 16-column sub-tiles' A-fragments first: two independent
            // WMMA accumulator chains for the scheduler to interleave
            v16h a0[4], a1[4];
#pragma unroll
            for (int kb = 0; kb < 4; ++kb) {
                const _Float16* p0 = F + (g * 32 +      l15) * DIM + kb * 32 + koff;
                const _Float16* p1 = F + (g * 32 + 16 + l15) * DIM + kb * 32 + koff;
                v8h lo0 = *reinterpret_cast<const v8h*>(p0);
                v8h hi0 = *reinterpret_cast<const v8h*>(p0 + 16);
                v8h lo1 = *reinterpret_cast<const v8h*>(p1);
                v8h hi1 = *reinterpret_cast<const v8h*>(p1 + 16);
                a0[kb] = __builtin_shufflevector(lo0, hi0, 0,1,2,3,4,5,6,7,8,9,10,11,12,13,14,15);
                a1[kb] = __builtin_shufflevector(lo1, hi1, 0,1,2,3,4,5,6,7,8,9,10,11,12,13,14,15);
            }
            v8f acc0 = {0.f,0.f,0.f,0.f,0.f,0.f,0.f,0.f};
            v8f acc1 = {0.f,0.f,0.f,0.f,0.f,0.f,0.f,0.f};
#pragma unroll
            for (int kb = 0; kb < 4; ++kb) {
                acc0 = __builtin_amdgcn_wmma_f32_16x16x32_f16(
                    false, a0[kb], false, q[kb], (short)0, acc0, false, false);
                acc1 = __builtin_amdgcn_wmma_f32_16x16x32_f16(
                    false, a1[kb], false, q[kb], (short)0, acc1, false, false);
            }
            // bounded scores (|s| <= 2/TEMP): exp(2x) = exp2(x * 2log2e).
            // raw v_exp_f32 (no libm range guard) is exact for our domain
            v4u pk[2], opk[2];
            {
                union { v4u u; _Float16 h[8]; } t0, t1;
#pragma unroll
                for (int r = 0; r < 8; ++r) {
                    t0.h[r] = (_Float16)__builtin_amdgcn_exp2f(acc0[r] * TWO_LOG2E);
                    t1.h[r] = (_Float16)__builtin_amdgcn_exp2f(acc1[r] * TWO_LOG2E);
                }
                pk[0] = t0.u; pk[1] = t1.u;
            }
#pragma unroll
            for (int s = 0; s < 2; ++s)
#pragma unroll
                for (int j2 = 0; j2 < 4; ++j2)
                    opk[s][j2] = (unsigned)__shfl_xor((int)pk[s][j2], 16);
            // assemble P^T as WMMA B-operand: half-lane swap of packed dwords
            union { v16h v; unsigned u[8]; } bb;
#pragma unroll
            for (int j2 = 0; j2 < 4; ++j2) {
                bb.u[j2]     = lo16 ? pk[0][j2]  : opk[1][j2];
                bb.u[4 + j2] = lo16 ? opk[0][j2] : pk[1][j2];
            }
            // V^T fragment (A-operand): lane row c = l15
            const _Float16* pv = Vt + l15 * CHUNK + g * 32 + (lo16 ? 0 : 16);
            v8h vlo = *reinterpret_cast<const v8h*>(pv);
            v8h vhi = *reinterpret_cast<const v8h*>(pv + 8);
            v16h vt = __builtin_shufflevector(vlo, vhi, 0,1,2,3,4,5,6,7,8,9,10,11,12,13,14,15);
            oacc = __builtin_amdgcn_wmma_f32_16x16x32_f16(
                false, vt, false, bb.v, (short)0, oacc, false, false);
        }
        __syncthreads();    // protect this buffer before it is re-filled next iteration
    }

    // O^T layout: lane = output row m; VGPR r -> class c = r (+8 for lanes>=16).
    // c == 10 is the softmax denominator (ones column).
    float c10    = oacc[2];
    float other  = __shfl_xor(c10, 16);
    float rowsum = lo16 ? other : c10;
    float inv    = 1.0f / rowsum;
    const int cbase = lo16 ? 0 : 8;
#pragma unroll
    for (int r = 0; r < 8; ++r) {
        int c = cbase + r;
        if (c < NCLS) out[(size_t)mRow * NCLS + c] = oacc[r] * inv * den[c];
    }
}

// ---------------------------------------------------------------- launch
extern "C" void kernel_launch(void* const* d_in, const int* in_sizes, int n_in,
                              void* d_out, int out_size, void* d_ws, size_t ws_size,
                              hipStream_t stream) {
    const float* anchor   = (const float*)d_in[0];
    const float* positive = (const float*)d_in[1];
    const float* lb_feat  = (const float*)d_in[2];
    const float* lb_oh    = (const float*)d_in[3];
    const float* lg1      = (const float*)d_in[4];
    const float* lg2      = (const float*)d_in[5];
    char* ws = (char*)d_ws;
    _Float16* feats = (_Float16*)(ws + OFF_FEATS);
    _Float16* VT    = (_Float16*)(ws + OFF_VT);
    float*    cnt   = (float*)(ws + OFF_CNT);
    float*    den   = (float*)(ws + OFF_DEN);
    float*    out   = (float*)d_out;

    init_kernel<<<1, 32, 0, stream>>>(cnt);
    prep_lb_kernel<<<NB_LB / 256, 256, 0, stream>>>(lb_oh, VT, cnt);
    prep_ulb_kernel<<<NB_ULB / 256, 256, 0, stream>>>(lg1, lg2, VT, cnt);
    norm_kernel<<<N_TOTAL / 8, 256, 0, stream>>>(lb_feat, anchor, positive, feats);
    finalize_kernel<<<1, 32, 0, stream>>>(cnt, den);
    attn_kernel<<<N_TOTAL / 64, 128, 0, stream>>>(feats, VT, den, out);
}
